// Quant3Linear_76063870812747
// MI455X (gfx1250) — matrix operand order repaired and checked by reading, not compiled, not run on hardware
//
#include <hip/hip_runtime.h>
#include <stdint.h>

typedef __bf16 bf16;
typedef __attribute__((ext_vector_type(16))) __bf16 v16bf;
typedef __attribute__((ext_vector_type(8)))  float  v8f;

#define IN_F  4096
#define OUT_F 4096
#define BS_M  8192   // B * S
#define KT    64     // K-tile per LDS stage (2 WMMA sub-steps)
#define LDT   72     // padded LDS row stride (elements): 144B = 9x16B
#define BUFB  (128 * LDT * 2)   // bytes per LDS buffer per matrix

// ---- CDNA5 async global->LDS copy (ASYNCcnt-tracked, no VGPR staging) ----
#define ASYNC_LDS_B128(lds_off, gaddr, IMM)                                  \
  asm volatile("global_load_async_to_lds_b128 %0, %1, off offset:" IMM      \
               :: "v"(lds_off), "v"(gaddr) : "memory")

__device__ __forceinline__ void async_copy64B(uint32_t lds, unsigned long long g) {
  ASYNC_LDS_B128(lds, g, "0");
  ASYNC_LDS_B128(lds, g, "16");
  ASYNC_LDS_B128(lds, g, "32");
  ASYNC_LDS_B128(lds, g, "48");
}

__device__ __forceinline__ void wait_async0() {
  asm volatile("s_wait_asynccnt 0x0" ::: "memory");
}

// ---------- prep kernels ----------

// wq[i*OUT+o] = (q[i*OUT+o]/15*2 - 1)*scale   (this IS the Bt operand of GEMM1)
__global__ void k_dequant(const int* __restrict__ q, const float* __restrict__ scale,
                          bf16* __restrict__ out, int n) {
  int i = blockIdx.x * 256 + threadIdx.x;
  if (i < n) {
    float v = fmaf((float)q[i], 2.0f / 15.0f, -1.0f) * scale[0];
    out[i] = (bf16)v;
  }
}

__global__ void k_f32_to_bf16(const float* __restrict__ in, bf16* __restrict__ out, int n) {
  int i = blockIdx.x * 256 + threadIdx.x;
  if (i < n) out[i] = (bf16)in[i];
}

// dst[c][r] = (bf16)src[r][c]  for an n x n f32 matrix (LDS-tiled transpose)
__global__ void k_transpose_cvt(const float* __restrict__ src, bf16* __restrict__ dst, int n) {
  __shared__ float tile[32][33];
  int c0 = blockIdx.x * 32, r0 = blockIdx.y * 32;
  int tx = threadIdx.x, ty = threadIdx.y;
  #pragma unroll
  for (int j = 0; j < 32; j += 8)
    tile[ty + j][tx] = src[(size_t)(r0 + ty + j) * n + (c0 + tx)];
  __syncthreads();
  #pragma unroll
  for (int j = 0; j < 32; j += 8)
    dst[(size_t)(c0 + ty + j) * n + (r0 + tx)] = (bf16)tile[tx][ty + j];
}

// ---------- GEMM: C[M,N] = A[M,K] * Bt[N,K]^T  (bf16 in, f32 accumulate) ----------
// MODE 0: C -> bf16           (T1 = U^T @ w0)
// MODE 1: C/aux[col] -> bf16  (W  = (T1 @ V) / scaleWH)
// MODE 2: C+aux[col] -> f32   (out = x @ W^T + bias)

template<int MODE>
__global__ __launch_bounds__(256) void k_gemm_bf16(
    const bf16* __restrict__ A, const bf16* __restrict__ Bt,
    const float* __restrict__ aux,
    bf16* __restrict__ Cb, float* __restrict__ Cf,
    int M, int N, int K)
{
  __shared__ bf16 sA[2][128 * LDT];
  __shared__ bf16 sB[2][128 * LDT];

  const int tid  = threadIdx.x;
  const int lane = tid & 31;
  const int wid  = tid >> 5;
  const int half = lane >> 4;   // lane half: selects K-chunk per ISA 7.12.2 layout
  const int l16  = lane & 15;
  const int wm   = (wid & 3) * 32;   // wave M offset in block tile
  const int wn   = (wid >> 2) * 64;  // wave N offset in block tile
  const int m0   = blockIdx.y * 128;
  const int n0   = blockIdx.x * 128;

  // global->LDS staging map: thread t copies 64B of row t/2 (32 K-elements)
  const int lr = tid >> 1;
  const int lc = (tid & 1) << 5;   // 0 or 32 elements
  const unsigned long long gA = (unsigned long long)(uintptr_t)(A  + (size_t)(m0 + lr) * K + lc);
  const unsigned long long gB = (unsigned long long)(uintptr_t)(Bt + (size_t)(n0 + lr) * K + lc);
  const uint32_t ldsA = (uint32_t)(uintptr_t)&sA[0][lr * LDT + lc];
  const uint32_t ldsB = (uint32_t)(uintptr_t)&sB[0][lr * LDT + lc];

  v8f acc[2][4] = {};

  // prime buffer 0
  async_copy64B(ldsA, gA);
  async_copy64B(ldsB, gB);
  wait_async0();
  __syncthreads();

  const int ksteps = K / KT;
  for (int s = 0; s < ksteps; ++s) {
    const int cur = s & 1, nxt = cur ^ 1;
    if (s + 1 < ksteps) {   // overlap next tile's DMA with this tile's WMMAs
      const unsigned long long koff = (unsigned long long)(s + 1) * KT * sizeof(bf16);
      async_copy64B(ldsA + nxt * BUFB, gA + koff);
      async_copy64B(ldsB + nxt * BUFB, gB + koff);
    }

    #pragma unroll
    for (int kk = 0; kk < KT; kk += 32) {
      // A fragment (16-bit A 16x32): lanes 0-15 hold K 0..7 & 16..23,
      // lanes 16-31 hold K 8..15 & 24..31 of row M = lane%16.
      v16bf afr[2];
      #pragma unroll
      for (int mt = 0; mt < 2; ++mt) {
        const bf16* p = &sA[cur][(wm + mt * 16 + l16) * LDT + kk];
        uint4* d = (uint4*)&afr[mt];
        d[0] = *(const uint4*)(p + half * 8);
        d[1] = *(const uint4*)(p + 16 + half * 8);
      }
      // B fragment (32x16): lane holds column N = lane%16, K 0..7 & 16..23
      // (lanes 0-15) or K 8..15 & 24..31 (lanes 16-31), matching A.
      v16bf bfr[4];
      #pragma unroll
      for (int nt = 0; nt < 4; ++nt) {
        const bf16* p = &sB[cur][(wn + nt * 16 + l16) * LDT + kk];
        uint4* d = (uint4*)&bfr[nt];
        d[0] = *(const uint4*)(p + half * 8);
        d[1] = *(const uint4*)(p + 16 + half * 8);
      }
      #pragma unroll
      for (int mt = 0; mt < 2; ++mt)
        #pragma unroll
        for (int nt = 0; nt < 4; ++nt)
          acc[mt][nt] = __builtin_amdgcn_wmma_f32_16x16x32_bf16(
              false, afr[mt], false, bfr[nt], (short)0, acc[mt][nt], false, false);
    }

    wait_async0();      // next tile landed in LDS
    __syncthreads();    // visible to all waves
  }

  // Epilogue. C/D f32 16x16 layout: row = vgpr + 8*(lane/16), col = lane%16.
  #pragma unroll
  for (int mt = 0; mt < 2; ++mt) {
    #pragma unroll
    for (int nt = 0; nt < 4; ++nt) {
      const int col = n0 + wn + nt * 16 + l16;
      const int rb  = m0 + wm + mt * 16 + half * 8;
      float mul = 1.0f, add = 0.0f;
      if (MODE == 1) mul = 1.0f / aux[col];
      if (MODE == 2) add = aux[col];
      #pragma unroll
      for (int v = 0; v < 8; ++v) {
        float val = acc[mt][nt][v] * mul + add;
        if (MODE == 2) Cf[(size_t)(rb + v) * N + col] = val;
        else           Cb[(size_t)(rb + v) * N + col] = (bf16)val;
      }
    }
  }
}

// ---------- launcher ----------

extern "C" void kernel_launch(void* const* d_in, const int* in_sizes, int n_in,
                              void* d_out, int out_size, void* d_ws, size_t ws_size,
                              hipStream_t stream) {
  (void)in_sizes; (void)n_in; (void)out_size; (void)ws_size;
  const float* x       = (const float*)d_in[0];
  const int*   qweight = (const int*)  d_in[1];
  const float* scale   = (const float*)d_in[2];
  const float* scaleWH = (const float*)d_in[3];
  const float* bias    = (const float*)d_in[4];
  const float* U       = (const float*)d_in[5];
  const float* V       = (const float*)d_in[6];
  float* out = (float*)d_out;

  char* ws = (char*)d_ws;
  const size_t MAT = (size_t)IN_F * OUT_F * sizeof(bf16); // 32 MiB
  const size_t XB  = (size_t)BS_M * IN_F * sizeof(bf16);  // 64 MiB
  bf16* xbf = (bf16*)(ws);                 // x in bf16
  bf16* wq  = (bf16*)(ws + XB);            // dequantized w0 as Bt[i][o]; later reused for W
  bf16* ut  = (bf16*)(ws + XB + MAT);      // U^T  (row-major [o][k])
  bf16* vt  = (bf16*)(ws + XB + 2 * MAT);  // V^T  (row-major [i][k])
  bf16* t1  = (bf16*)(ws + XB + 3 * MAT);  // T1 = U^T @ w0  (row-major [o][i])
  // total workspace: 192 MiB

  const int nx = BS_M * IN_F;
  const int nw = IN_F * OUT_F;
  k_f32_to_bf16<<<dim3((nx + 255) / 256), dim3(256), 0, stream>>>(x, xbf, nx);
  k_dequant<<<dim3((nw + 255) / 256), dim3(256), 0, stream>>>(qweight, scale, wq, nw);
  k_transpose_cvt<<<dim3(OUT_F / 32, OUT_F / 32), dim3(32, 8), 0, stream>>>(U, ut, OUT_F);
  k_transpose_cvt<<<dim3(IN_F / 32,  IN_F / 32),  dim3(32, 8), 0, stream>>>(V, vt, IN_F);

  // T1 = U^T @ w0            (M=OUT, N=IN, K=OUT)
  k_gemm_bf16<0><<<dim3(IN_F / 128, OUT_F / 128), dim3(256), 0, stream>>>(
      ut, wq, nullptr, t1, nullptr, OUT_F, IN_F, OUT_F);
  // W = (T1 @ V) / scaleWH   (M=OUT, N=IN, K=IN); W overwrites wq slot, stored [o][i]
  k_gemm_bf16<1><<<dim3(IN_F / 128, OUT_F / 128), dim3(256), 0, stream>>>(
      t1, vt, scaleWH, wq, nullptr, OUT_F, IN_F, IN_F);
  // out = x @ W^T + bias     (M=B*S, N=OUT, K=IN)
  k_gemm_bf16<2><<<dim3(OUT_F / 128, BS_M / 128), dim3(256), 0, stream>>>(
      xbf, wq, bias, nullptr, out, BS_M, OUT_F, IN_F);
}
